// MultiHeadAttention_63153199120511
// MI455X (gfx1250) — matrix operand order repaired and checked by reading, not compiled, not run on hardware
//
#include <hip/hip_runtime.h>
#include <hip/hip_bf16.h>

// MHA forward for B=2,S=2048,D=1024,H=16,DK=64 on gfx1250 (CDNA5, wave32).
// All matmuls via v_wmma_f32_16x16x32_f16. attn (512MB) streamed with
// non-temporal stores. Projection GEMM double-buffers tiles in LDS:
//   X tile  (64x32 f32)  via GLOBAL_LOAD_ASYNC_TO_LDS_B128 (ASYNCcnt)
//   W tile (128x32 f32)  via TENSOR_LOAD_TO_LDS / TDM      (TENSORcnt)
// Workspace: Qh,Kh [B,H,S,DK] f16 + VhT [B,H,DK,S] f16 = 24MB.

#define BB  2
#define SS  2048
#define DD  1024
#define HH  16
#define DKK 64

typedef _Float16 h16 __attribute__((ext_vector_type(16)));
typedef _Float16 h8  __attribute__((ext_vector_type(8)));
typedef float    f8  __attribute__((ext_vector_type(8)));
typedef float    f4  __attribute__((ext_vector_type(4)));

__device__ __forceinline__ f8 wmma16(h16 a, h16 b, f8 c) {
  // D = A(16x32 f16) * B(32x16 f16) + C(16x16 f32)
  return __builtin_amdgcn_wmma_f32_16x16x32_f16(false, a, false, b,
                                                (short)0, c, false, false);
}

// ---- Async per-lane DMA: 16B global -> LDS, tracked by ASYNCcnt ----
// LDS address = low 32 bits of the generic pointer (flat LDS aperture
// truncates to the wave-relative LDS offset, which is what VDST wants).
__device__ __forceinline__ void async_copy_b128(void* lds, const void* g) {
  unsigned l = (unsigned)(size_t)lds;
  unsigned long long ga = (unsigned long long)(size_t)g;
  asm volatile("global_load_async_to_lds_b128 %0, %1, off"
               :: "v"(l), "v"(ga) : "memory");
}
__device__ __forceinline__ void wait_async0() {
  asm volatile("s_wait_asynccnt 0x0" ::: "memory");
}

// ---- Tensor Data Mover: 2-D f32 tile global -> LDS, tracked by TENSORcnt ---
#if __has_builtin(__builtin_amdgcn_tensor_load_to_lds)
#define HAVE_TDM 1
typedef unsigned u32x4 __attribute__((ext_vector_type(4)));
typedef int      i32x4 __attribute__((ext_vector_type(4)));
typedef int      i32x8 __attribute__((ext_vector_type(8)));

// D# per CDNA5 ISA ch.8: group0 = {flags, lds_addr, global_addr, type},
// group1 = {mask/data_size, tensor_dim0/1, tile_dim0/1/2, dim0_stride}.
__device__ __forceinline__ void tdm_load_2d_f32(void* lds, const void* g,
                                                unsigned tensorDim0,
                                                unsigned tensorDim1,
                                                unsigned tileDim0,
                                                unsigned tileDim1,
                                                unsigned stride0) {
  unsigned long long ga = (unsigned long long)(size_t)g;
  u32x4 g0;
  g0[0] = 1u;                                            // count=1, user load D#
  g0[1] = (unsigned)(size_t)lds;                         // lds_addr (bytes)
  g0[2] = (unsigned)(ga & 0xFFFFFFFFull);                // global_addr[31:0]
  g0[3] = (unsigned)((ga >> 32) & 0x01FFFFFFu) | (2u << 30); // [56:32] | type=2
  i32x8 g1;
  g1[0] = (int)(2u << 16);                               // data_size = 4B
  g1[1] = (int)((tensorDim0 & 0xFFFFu) << 16);           // dim0[15:0] @ bits63:48
  g1[2] = (int)(((tensorDim1 & 0xFFFFu) << 16) | ((tensorDim0 >> 16) & 0xFFFFu));
  g1[3] = (int)(((tileDim0 & 0xFFFFu) << 16) | ((tensorDim1 >> 16) & 0xFFFFu));
  g1[4] = (int)(tileDim1 & 0xFFFFu);                     // tile_dim1, tile_dim2=0
  g1[5] = (int)stride0;                                  // dim0_stride[31:0]
  g1[6] = 0;                                             // stride hi bits
  g1[7] = 0;
  i32x4 zz = {0, 0, 0, 0};
#if __clang_major__ >= 23
  i32x8 z8 = {0, 0, 0, 0, 0, 0, 0, 0};
  __builtin_amdgcn_tensor_load_to_lds(g0, g1, zz, zz, z8, 0);
#else
  __builtin_amdgcn_tensor_load_to_lds(g0, g1, zz, zz, 0);
#endif
}
#endif

// A-fragment (16x32 f16) from f32 source: f0..f3 packed in order, k order set by caller
// (proj: k = hi*16 + {0..15}; P@V: k = hi*8 + {0..7} and hi*8 + 16 + {0..7})  (hi = lane>=16).
__device__ __forceinline__ h16 pack_a_f32(f4 f0, f4 f1, f4 f2, f4 f3) {
  h16 v;
#pragma unroll
  for (int j = 0; j < 4; ++j) {
    v[j]      = (_Float16)f0[j];
    v[4 + j]  = (_Float16)f1[j];
    v[8 + j]  = (_Float16)f2[j];
    v[12 + j] = (_Float16)f3[j];
  }
  return v;
}

// B-fragment (32x16 f16) from f32 source: per-lane 16 contiguous along K
// (k = (lane>=16)*16 + j), column n = lane%16.
__device__ __forceinline__ h16 pack_b_f32(const float* p) {
  f4 w0 = *(const f4*)p;
  f4 w1 = *(const f4*)(p + 4);
  f4 w2 = *(const f4*)(p + 8);
  f4 w3 = *(const f4*)(p + 12);
  h16 v;
#pragma unroll
  for (int j = 0; j < 4; ++j) {
    v[j]      = (_Float16)w0[j];
    v[4 + j]  = (_Float16)w1[j];
    v[8 + j]  = (_Float16)w2[j];
    v[12 + j] = (_Float16)w3[j];
  }
  return v;
}

__device__ __forceinline__ h16 cat_h8(h8 a, h8 b) {
  h16 v;
#pragma unroll
  for (int j = 0; j < 8; ++j) { v[j] = a[j]; v[8 + j] = b[j]; }
  return v;
}

// ---------------------------------------------------------------------------
// Projection: Y = X @ W^T, cast to f16.
// X: [B*S, D] f32 row-major. W: [D, D] f32 ([out,in]; contraction along rows).
// TRANS==0: Y[b][h][s][dk] (dk contiguous)  -- Q, K
// TRANS==1: Y[b][h][dk][s] (s contiguous)   -- V
// Block: 128 threads (4 waves). Block tile 64(m) x 128(n); wave tile 32x64.
// LDS double buffer: xs[2][64][32] f32 (16KB, async) + wt[2][128][32] f32
// (32KB, TDM). Each element fetched from global ONCE per block.
// ---------------------------------------------------------------------------
template <int TRANS>
__global__ __launch_bounds__(128) void proj_kernel(
    const float* __restrict__ X, const float* __restrict__ W,
    _Float16* __restrict__ Y) {
  __shared__ float xs[2][64][32];
  __shared__ float wt[2][128][32];

  const int tid      = threadIdx.x;
  const int lane     = tid & 31;
  const int wave     = tid >> 5;
  const int laneHalf = lane & 15;
  const int hi       = lane >> 4;
  const int m0blk = blockIdx.x * 64;
  const int n0blk = blockIdx.y * 128;
  const int mW = (wave >> 1) * 32;   // wave's m offset within block tile
  const int nW = (wave & 1) * 64;    // wave's n offset within block tile

  f8 acc[2][4];
#pragma unroll
  for (int mt = 0; mt < 2; ++mt)
#pragma unroll
    for (int nt = 0; nt < 4; ++nt)
#pragma unroll
      for (int r = 0; r < 8; ++r) acc[mt][nt][r] = 0.f;

  // X tile: 512 x 16B chunks, per-lane async copies (4 per thread).
  auto stageX = [&](int buf, int k0) {
#pragma unroll
    for (int i = 0; i < 4; ++i) {
      int c = tid + i * 128;
      int row = c >> 3, col = (c & 7) * 4;
      async_copy_b128(&xs[buf][row][col],
                      X + (size_t)(m0blk + row) * DD + k0 + col);
    }
  };
  // W tile: one TDM descriptor moves the whole 128x32 f32 tile.
  auto stageW = [&](int buf, int k0) {
#if HAVE_TDM
    if (tid == 0)   // tensor ops ignore EXEC; one lane -> exactly one DMA
      tdm_load_2d_f32(&wt[buf][0][0], W + (size_t)n0blk * DD + k0,
                      DD, DD, 32, 128, DD);
#else
#pragma unroll
    for (int i = 0; i < 8; ++i) {
      int c = tid + i * 128;
      int row = c >> 3, col = (c & 7) * 4;
      async_copy_b128(&wt[buf][row][col],
                      W + (size_t)(n0blk + row) * DD + k0 + col);
    }
#endif
  };
  auto waitCopies = [&]() {
    wait_async0();
#if HAVE_TDM
    __builtin_amdgcn_s_wait_tensorcnt(0);   // no-op for non-issuing waves
#endif
  };

  stageX(0, 0);
  stageW(0, 0);
  waitCopies();
  __syncthreads();

  int cur = 0;
  for (int k0 = 0; k0 < DD; k0 += 32) {
    if (k0 + 32 < DD) {                       // overlap next copy with compute
      stageX(cur ^ 1, k0 + 32);
      stageW(cur ^ 1, k0 + 32);
    }

    h16 a[2];
#pragma unroll
    for (int mt = 0; mt < 2; ++mt) {
      const float* ap = &xs[cur][mW + mt * 16 + laneHalf][hi * 16];
      f4 f0 = *(const f4*)ap;
      f4 f1 = *(const f4*)(ap + 4);
      f4 f2 = *(const f4*)(ap + 8);
      f4 f3 = *(const f4*)(ap + 12);
      a[mt] = pack_a_f32(f0, f1, f2, f3);
    }
    h16 bfr[4];
#pragma unroll
    for (int nt = 0; nt < 4; ++nt)
      bfr[nt] = pack_b_f32(&wt[cur][nW + nt * 16 + laneHalf][hi * 16]);
#pragma unroll
    for (int mt = 0; mt < 2; ++mt)
#pragma unroll
      for (int nt = 0; nt < 4; ++nt)
        acc[mt][nt] = wmma16(a[mt], bfr[nt], acc[mt][nt]);

    waitCopies();        // next buffer fully written
    __syncthreads();     // everyone done reading cur / writing next
    cur ^= 1;
  }

#pragma unroll
  for (int mt = 0; mt < 2; ++mt) {
#pragma unroll
    for (int nt = 0; nt < 4; ++nt) {
      const int n    = n0blk + nW + nt * 16 + laneHalf;  // feature 0..1023
      const int hh   = n >> 6;
      const int dk   = n & 63;
      const int mrow = m0blk + mW + mt * 16 + hi * 8;    // 8 rows, aligned
      const int bb   = mrow >> 11;
      const int s0   = mrow & (SS - 1);
      if (TRANS) {
        h8 hv;
#pragma unroll
        for (int r = 0; r < 8; ++r) hv[r] = (_Float16)acc[mt][nt][r];
        _Float16* yp = Y + (((size_t)bb * HH + hh) * DKK + dk) * SS + s0;
        *(h8*)yp = hv;                            // 16B contiguous along s
      } else {
        _Float16* yp = Y + (((size_t)bb * HH + hh) * SS + s0) * DKK + dk;
#pragma unroll
        for (int r = 0; r < 8; ++r) yp[(size_t)r * DKK] = (_Float16)acc[mt][nt][r];
      }
    }
  }
}

// ---------------------------------------------------------------------------
// Fused attention: per block = one (b,h) and one 16-row q-tile.
// Phase 1: scores(16x2048) = Q Kt * 0.125 -> LDS (WMMA), K rows prefetched.
// Phase 2: row softmax in LDS; normalized probs streamed to attn (d_out).
// Phase 3: out(16x64) = P @ V (A from LDS f32->f16, B from VhT), streamed out.
// 128 threads (4 waves), 128 KB LDS.
// ---------------------------------------------------------------------------
__global__ __launch_bounds__(128) void attn_kernel(
    const _Float16* __restrict__ Qh, const _Float16* __restrict__ Kh,
    const _Float16* __restrict__ VhT, float* __restrict__ out,
    float* __restrict__ attn) {
  __shared__ float sc[16 * SS];                   // 131072 bytes

  const int lane     = threadIdx.x & 31;
  const int wave     = threadIdx.x >> 5;
  const int laneHalf = lane & 15;
  const int hi       = lane >> 4;
  const int bh = blockIdx.x >> 7;                 // 0..31  (b*H + h)
  const int q0 = (blockIdx.x & 127) * 16;
  const int b  = bh >> 4;
  const int h  = bh & 15;

  const _Float16* Qp = Qh + ((size_t)bh * SS + q0) * DKK;
  const _Float16* Kp = Kh + (size_t)bh * SS * DKK;
  const _Float16* Vp = VhT + (size_t)bh * DKK * SS;
  float* attnP = attn + ((size_t)bh * SS + q0) * SS;
  float* outP  = out + ((size_t)b * SS + q0) * DD + h * DKK;

  // ---- Phase 1: scores -> LDS ----
  h16 aq[2];
#pragma unroll
  for (int kt = 0; kt < 2; ++kt) {
    const _Float16* ap = Qp + (size_t)laneHalf * DKK + kt * 32 + hi * 8;
    h8 r0 = *(const h8*)ap;
    h8 r1 = *(const h8*)(ap + 16);
    aq[kt] = cat_h8(r0, r1);
  }
  for (int nt = wave; nt < SS / 16; nt += 4) {    // 32 key-tiles per wave
    const int key = nt * 16 + laneHalf;
    if (nt + 8 < SS / 16)                         // prefetch 2 tiles ahead
      __builtin_prefetch(Kp + (size_t)(key + 128) * DKK, 0, 3);
    f8 acc = {0.f, 0.f, 0.f, 0.f, 0.f, 0.f, 0.f, 0.f};
#pragma unroll
    for (int kt = 0; kt < 2; ++kt) {
      const _Float16* bp = Kp + (size_t)key * DKK + kt * 32 + hi * 8;
      h8 b0 = *(const h8*)bp;
      h8 b1 = *(const h8*)(bp + 16);
      acc = wmma16(aq[kt], cat_h8(b0, b1), acc);
    }
#pragma unroll
    for (int r = 0; r < 8; ++r)
      sc[(size_t)(r + 8 * hi) * SS + nt * 16 + laneHalf] = acc[r] * 0.125f;
  }
  __syncthreads();

  // ---- Phase 2: softmax per row + stream normalized attn ----
  for (int row = wave * 4; row < wave * 4 + 4; ++row) {
    float* rp = &sc[(size_t)row * SS];
    float mx = -3.0e38f;
    for (int i = lane; i < SS; i += 32) mx = fmaxf(mx, rp[i]);
#pragma unroll
    for (int o = 16; o > 0; o >>= 1) mx = fmaxf(mx, __shfl_xor(mx, o, 32));
    float sum = 0.f;
    for (int i = lane; i < SS; i += 32) {
      float e = __expf(rp[i] - mx);
      rp[i] = e;
      sum += e;
    }
#pragma unroll
    for (int o = 16; o > 0; o >>= 1) sum += __shfl_xor(sum, o, 32);
    const float inv = 1.f / sum;
    float* gp = attnP + (size_t)row * SS;
    for (int i = lane; i < SS; i += 32) {
      float p = rp[i] * inv;
      rp[i] = p;
      __builtin_nontemporal_store(p, gp + i);     // stream 512MB past L2
    }
  }
  __syncthreads();

  // ---- Phase 3: out = P @ V ----
  {
    const int dk = wave * 16 + laneHalf;          // one 16-col dk-tile per wave
    const _Float16* vrow = Vp + (size_t)dk * SS;  // VhT: contiguous along keys
    f8 acc = {0.f, 0.f, 0.f, 0.f, 0.f, 0.f, 0.f, 0.f};
    for (int kk = 0; kk < SS; kk += 32) {
      const float* ap = &sc[(size_t)laneHalf * SS + kk + hi * 8];
      f4 f0 = *(const f4*)ap;
      f4 f1 = *(const f4*)(ap + 4);
      f4 f2 = *(const f4*)(ap + 16);
      f4 f3 = *(const f4*)(ap + 20);
      h8 b0 = *(const h8*)(vrow + kk + hi * 8);
      h8 b1 = *(const h8*)(vrow + kk + hi * 8 + 16);
      acc = wmma16(pack_a_f32(f0, f1, f2, f3), cat_h8(b0, b1), acc);
    }
#pragma unroll
    for (int r = 0; r < 8; ++r)
      __builtin_nontemporal_store(acc[r], outP + (size_t)(r + 8 * hi) * DD + dk);
  }
}

// ---------------------------------------------------------------------------
extern "C" void kernel_launch(void* const* d_in, const int* in_sizes, int n_in,
                              void* d_out, int out_size, void* d_ws,
                              size_t ws_size, hipStream_t stream) {
  (void)in_sizes; (void)n_in; (void)out_size; (void)ws_size;
  const float* q  = (const float*)d_in[0];
  const float* k  = (const float*)d_in[1];
  const float* v  = (const float*)d_in[2];
  const float* Wq = (const float*)d_in[3];
  const float* Wk = (const float*)d_in[4];
  const float* Wv = (const float*)d_in[5];

  const size_t headElems = (size_t)BB * HH * SS * DKK;  // 4M f16 each
  _Float16* Qh  = (_Float16*)d_ws;
  _Float16* Kh  = Qh + headElems;
  _Float16* VhT = Kh + headElems;

  float* out  = (float*)d_out;
  float* attn = out + (size_t)BB * SS * DD;

  dim3 pg(BB * SS / 64, DD / 128);
  dim3 pb(128);
  proj_kernel<0><<<pg, pb, 0, stream>>>(q, Wq, Qh);
  proj_kernel<0><<<pg, pb, 0, stream>>>(k, Wk, Kh);
  proj_kernel<1><<<pg, pb, 0, stream>>>(v, Wv, VhT);

  attn_kernel<<<dim3(BB * HH * (SS / 16)), dim3(128), 0, stream>>>(
      Qh, Kh, VhT, out, attn);
}
